// ONLSTM_Cell_28372553957857
// MI455X (gfx1250) — matrix operand order repaired and checked by reading, not compiled, not run on hardware
//
#include <hip/hip_runtime.h>

// ---------------------------------------------------------------------------
// ON-LSTM cell for MI455X (gfx1250, wave32, WMMA).
//   pre = [inputs|hidden] @ [W_all;U_all] + b_all           (bf16 WMMA, f32 acc)
//   f,i,o = sigmoid; c_hat = tanh; f~ = cumsum_B(softmax(zft));
//   i~ = 1 - cumsum_B(softmax(zit)); gate combine; out = (hidden, cell)
// ---------------------------------------------------------------------------

typedef __attribute__((ext_vector_type(16))) __bf16 v16bf;
typedef __attribute__((ext_vector_type(8)))  __bf16 v8bf;
typedef __attribute__((ext_vector_type(8)))  float  v8f;

#define B_DIM 4096
#define U_DIM 1024
#define K_DIM 2048   // D + U
#define N_DIM 6144   // 6*U

static __device__ __forceinline__ unsigned short f32_to_bf16(float f) {
    union { float f; unsigned int u; } v; v.f = f;
    unsigned int u = v.u;
    u += 0x7FFFu + ((u >> 16) & 1u);   // round to nearest even
    return (unsigned short)(u >> 16);
}

// --------------------------- packing kernels -------------------------------

__global__ void pack_a_kernel(const float* __restrict__ inp,
                              const float* __restrict__ hid,
                              unsigned short* __restrict__ A) {
    int idx = blockIdx.x * 256 + threadIdx.x;         // B_DIM*K_DIM threads
    int b = idx >> 11;
    int k = idx & 2047;
    float v = (k < 1024) ? inp[(size_t)b * 1024 + k]
                         : hid[(size_t)b * 1024 + (k - 1024)];
    A[idx] = f32_to_bf16(v);
}

struct WPtrs { const float* W[6]; const float* U[6]; };
struct BPtrs { const float* b[6]; };

// Tiled transpose: Wt[n][k] (bf16, n-major) from column-stacked W_j / U_j.
__global__ void pack_w_kernel(WPtrs p, unsigned short* __restrict__ Wt) {
    __shared__ unsigned short tile[32][33];
    int g  = blockIdx.z;            // gate 0..5
    int n0 = blockIdx.y * 32;       // 0..1023
    int k0 = blockIdx.x * 32;       // 0..2047
    int tx = threadIdx.x & 31;
    int ty = threadIdx.x >> 5;      // 0..7
    const float* src;
    int kbase;
    if (k0 < 1024) { src = p.W[g]; kbase = k0; }
    else           { src = p.U[g]; kbase = k0 - 1024; }
#pragma unroll
    for (int yy = 0; yy < 4; ++yy) {
        int row = ty + yy * 8;
        tile[row][tx] = f32_to_bf16(src[(size_t)(kbase + row) * 1024 + n0 + tx]);
    }
    __syncthreads();
#pragma unroll
    for (int yy = 0; yy < 4; ++yy) {
        int row = ty + yy * 8;
        Wt[(size_t)(g * 1024 + n0 + row) * K_DIM + k0 + tx] = tile[tx][row];
    }
}

__global__ void pack_bias_kernel(BPtrs p, float* __restrict__ ball) {
    int n = blockIdx.x * 256 + threadIdx.x;           // N_DIM threads
    ball[n] = p.b[n >> 10][n & 1023];
}

// ------------------------------ WMMA GEMM ----------------------------------
// Block: 256 threads = 8 waves (2 M-waves x 4 N-waves). Wave tile: 64x64.
// Block tile: 128 (M) x 256 (N). Grid: (N/256=24, M/128=32). K = 2048.

__global__ __launch_bounds__(256, 1)
void gemm_wmma_kernel(const unsigned short* __restrict__ A,
                      const unsigned short* __restrict__ Wt,
                      const float* __restrict__ ball,
                      float* __restrict__ pre) {
    const int lane = threadIdx.x & 31;
    const int wave = threadIdx.x >> 5;
    const int wm   = wave >> 2;           // 0..1
    const int wn   = wave & 3;            // 0..3
    const int half = lane >> 4;           // 0..1
    const int l16  = lane & 15;           // 0..15
    const int m0 = blockIdx.y * 128 + wm * 64;
    const int n0 = blockIdx.x * 256 + wn * 64;

    v8f acc[4][4];
#pragma unroll
    for (int i = 0; i < 4; ++i)
#pragma unroll
        for (int j = 0; j < 4; ++j)
            acc[i][j] = (v8f){0.f, 0.f, 0.f, 0.f, 0.f, 0.f, 0.f, 0.f};

    // A frag: lane (m=l16, half) holds K = 16*half..+15 contiguous.
    const unsigned short* aptr[4];
#pragma unroll
    for (int i = 0; i < 4; ++i)
        aptr[i] = A + (size_t)(m0 + i * 16 + l16) * K_DIM + half * 16;
    // B frag: lane (n=l16, half) holds K = 16*half..+15 contiguous (Wt n-major).
    const unsigned short* bptr[4];
#pragma unroll
    for (int j = 0; j < 4; ++j)
        bptr[j] = Wt + (size_t)(n0 + j * 16 + l16) * K_DIM + half * 16;

    for (int k = 0; k < K_DIM; k += 32) {
        v16bf afrag[4], bfrag[4];
#pragma unroll
        for (int i = 0; i < 4; ++i) {
            v8bf lo = *(const v8bf*)(aptr[i] + k);
            v8bf hi = *(const v8bf*)(aptr[i] + k + 8);
            afrag[i] = __builtin_shufflevector(lo, hi,
                0, 1, 2, 3, 4, 5, 6, 7, 8, 9, 10, 11, 12, 13, 14, 15);
        }
#pragma unroll
        for (int j = 0; j < 4; ++j)
            bfrag[j] = *(const v16bf*)(bptr[j] + k);
#pragma unroll
        for (int i = 0; i < 4; ++i)
#pragma unroll
            for (int j = 0; j < 4; ++j)
                acc[i][j] = __builtin_amdgcn_wmma_f32_16x16x32_bf16(
                    false, afrag[i], false, bfrag[j],
                    (short)0, acc[i][j], false, false);
    }

    // Epilogue: D layout VGPR r -> row r + 8*half, col l16. Add bias, store.
#pragma unroll
    for (int j = 0; j < 4; ++j) {
        int n = n0 + j * 16 + l16;
        float bias = ball[n];
#pragma unroll
        for (int i = 0; i < 4; ++i) {
            int mrow = m0 + i * 16 + half * 8;
            float* out = pre + (size_t)mrow * N_DIM + n;
#pragma unroll
            for (int r = 0; r < 8; ++r)
                out[(size_t)r * N_DIM] = acc[i][j][r] + bias;
        }
    }
}

// ------------------------- softmax over last axis --------------------------
// One 256-thread block per (row, gate in {ft,it}); 1024 elems per row chunk.

__global__ void softmax_rows_kernel(float* __restrict__ pre) {
    __shared__ float red[256];
    int r = blockIdx.x >> 1;
    int g = blockIdx.x & 1;
    float* row = pre + (size_t)r * N_DIM + (4 + g) * 1024;
    int t = threadIdx.x;
    float x0 = row[t], x1 = row[t + 256], x2 = row[t + 512], x3 = row[t + 768];
    float mx = fmaxf(fmaxf(x0, x1), fmaxf(x2, x3));
    red[t] = mx; __syncthreads();
    for (int s = 128; s > 0; s >>= 1) {
        if (t < s) red[t] = fmaxf(red[t], red[t + s]);
        __syncthreads();
    }
    mx = red[0]; __syncthreads();
    float e0 = __expf(x0 - mx), e1 = __expf(x1 - mx);
    float e2 = __expf(x2 - mx), e3 = __expf(x3 - mx);
    red[t] = e0 + e1 + e2 + e3; __syncthreads();
    for (int s = 128; s > 0; s >>= 1) {
        if (t < s) red[t] += red[t + s];
        __syncthreads();
    }
    float inv = 1.0f / red[0];
    row[t] = e0 * inv; row[t + 256] = e1 * inv;
    row[t + 512] = e2 * inv; row[t + 768] = e3 * inv;
}

// ----------------------- cumsum over batch axis ----------------------------
// One thread per column (2 gates x 1024 cols); coalesced across lanes.

__global__ void cumsum_cols_kernel(float* __restrict__ pre) {
    int t = blockIdx.x * 256 + threadIdx.x;   // 2048 threads
    int g = t >> 10;
    int c = t & 1023;
    float* p = pre + (size_t)(4 + g) * 1024 + c;
    float acc = 0.f;
    for (int b = 0; b < B_DIM; ++b) {
        acc += p[(size_t)b * N_DIM];
        p[(size_t)b * N_DIM] = acc;
    }
}

// ------------------------------ epilogue -----------------------------------

__global__ void epilogue_kernel(const float* __restrict__ pre,
                                const float* __restrict__ cell_prev,
                                float* __restrict__ out) {
    int idx = blockIdx.x * 256 + threadIdx.x;   // B_DIM*U_DIM threads
    int b = idx >> 10;
    int u = idx & 1023;
    const float* row = pre + (size_t)b * N_DIM;
    float zf = row[u],        zi  = row[1024 + u];
    float zo = row[2048 + u], zc  = row[3072 + u];
    float ft = row[4096 + u], itc = row[5120 + u];
    float f  = 1.f / (1.f + __expf(-zf));
    float i  = 1.f / (1.f + __expf(-zi));
    float o  = 1.f / (1.f + __expf(-zo));
    float ch = tanhf(zc);
    float it = 1.f - itc;
    float omega = ft * it;
    float fh = f * omega + (ft - omega);
    float ih = i * omega + (it - omega);
    float cell = fh * cell_prev[idx] + ih * ch;
    float hidden = o * tanhf(cell);
    out[idx] = hidden;
    out[(size_t)B_DIM * U_DIM + idx] = cell;
}

// ------------------------------ launcher -----------------------------------

extern "C" void kernel_launch(void* const* d_in, const int* in_sizes, int n_in,
                              void* d_out, int out_size, void* d_ws, size_t ws_size,
                              hipStream_t stream) {
    (void)in_sizes; (void)n_in; (void)out_size; (void)ws_size;

    const float* inputs      = (const float*)d_in[0];
    const float* hidden_prev = (const float*)d_in[1];
    const float* cell_prev   = (const float*)d_in[2];

    WPtrs wp; BPtrs bp;
    for (int j = 0; j < 6; ++j) {
        wp.W[j] = (const float*)d_in[3 + 3 * j];
        wp.U[j] = (const float*)d_in[4 + 3 * j];
        bp.b[j] = (const float*)d_in[5 + 3 * j];
    }

    // Workspace layout (bytes, 256-aligned):
    //   A_bf16  [B,2048]      16,777,216
    //   Wt_bf16 [6144,2048]   25,165,824
    //   b_all   [6144] f32        24,576
    //   pre     [B,6144] f32 100,663,296
    char* ws = (char*)d_ws;
    unsigned short* A_bf  = (unsigned short*)(ws);
    unsigned short* Wt_bf = (unsigned short*)(ws + 16777216);
    float*          b_all = (float*)(ws + 16777216 + 25165824);
    float*          pre   = (float*)(ws + 16777216 + 25165824 + 24576);

    pack_a_kernel<<<(B_DIM * K_DIM) / 256, 256, 0, stream>>>(inputs, hidden_prev, A_bf);
    pack_w_kernel<<<dim3(K_DIM / 32, U_DIM / 32, 6), 256, 0, stream>>>(wp, Wt_bf);
    pack_bias_kernel<<<N_DIM / 256, 256, 0, stream>>>(bp, b_all);

    gemm_wmma_kernel<<<dim3(N_DIM / 256, B_DIM / 128), 256, 0, stream>>>(
        A_bf, Wt_bf, b_all, pre);

    softmax_rows_kernel<<<B_DIM * 2, 256, 0, stream>>>(pre);
    cumsum_cols_kernel<<<2048 / 256, 256, 0, stream>>>(pre);
    epilogue_kernel<<<(B_DIM * U_DIM) / 256, 256, 0, stream>>>(pre, cell_prev,
                                                               (float*)d_out);
}
